// AgnosticResidualInteractionBlock_34522947125576
// MI455X (gfx1250) — matrix operand order repaired and checked by reading, not compiled, not run on hardware
//
#include <hip/hip_runtime.h>
#include <math.h>

// Problem constants (match reference)
#define N_NODES 10000
#define N_EDGES 160000
#define MULC    64
#define LMD     9
#define NATT    10
#define NBAS    8
#define HIDD    64
#define NWGT    81
#define FEAT    (MULC * LMD)   // 576
#define HPAD    32             // K-padded l-stride of h workspace (WMMA K tile)

typedef __bf16 bf16;
typedef __attribute__((ext_vector_type(8)))  __bf16 bf16x8;
typedef __attribute__((ext_vector_type(16))) __bf16 bf16x16;
typedef __attribute__((ext_vector_type(8)))  float  f32x8;

// ---------------------------------------------------------------------------
// fragment helpers (generic pointers; addrspace resolved by inlining)
// ---------------------------------------------------------------------------

// A fragment: 16x32 bf16 tile, row-major, row stride `stride` (ISA 7.12.2):
// lanes 0-15 : row=lane,    K = {k0..k0+7} U {k0+16..k0+23}
// lanes 16-31: row=lane-16, K = {k0+8..k0+15} U {k0+24..k0+31}
__device__ __forceinline__ bf16x16 frag_A(const bf16* A, int stride, int k0, int lane) {
  const int row = lane & 15;
  const int kb  = k0 + ((lane >> 4) << 3);
  const bf16* base = A + row * stride + kb;
  bf16x8 lo = *(const bf16x8*)(base);
  bf16x8 hi = *(const bf16x8*)(base + 16);
  return __builtin_shufflevector(lo, hi, 0, 1, 2, 3, 4, 5, 6, 7,
                                 8, 9, 10, 11, 12, 13, 14, 15);
}

// B fragment from a column-major (transposed) tile BT[n][Kp]:
// lanes 0-15: col=n0+lane, K = {k0..k0+7} U {k0+16..k0+23} ; lanes 16-31: col=n0+lane-16, K = {k0+8..k0+15} U {k0+24..k0+31}.
// Per lane: two 16B runs -> 2 x b128 loads.
__device__ __forceinline__ bf16x16 frag_BT(const bf16* BT, int Kp, int k0, int n0, int lane) {
  const int col = (lane & 15) + n0;
  const int kb  = k0 + ((lane >> 4) << 3);
  const bf16* base = BT + (size_t)col * Kp + kb;
  bf16x8 lo = *(const bf16x8*)(base);
  bf16x8 hi = *(const bf16x8*)(base + 16);
  return __builtin_shufflevector(lo, hi, 0, 1, 2, 3, 4, 5, 6, 7,
                                 8, 9, 10, 11, 12, 13, 14, 15);
}

__device__ __forceinline__ f32x8 wmma_bf16(bf16x16 a, bf16x16 b, f32x8 c) {
  return __builtin_amdgcn_wmma_f32_16x16x32_bf16(false, a, false, b, (short)0, c, false, false);
}

__device__ __forceinline__ float silu(float v) { return v / (1.0f + __expf(-v)); }

// ---------------------------------------------------------------------------
// Prep kernel: transpose fp32 weight [K][N] (batched) into zero-padded bf16
// BT[b][Np][Kp]. Runs once per launch; guards are fine here.
// ---------------------------------------------------------------------------
__global__ void transpose_pad_bf16(const float* __restrict__ src, bf16* __restrict__ dst,
                                   int K, int N, int Kp, int Np) {
  const int total = Kp * Np;
  const int b = blockIdx.y;
  for (int idx = blockIdx.x * blockDim.x + threadIdx.x; idx < total;
       idx += gridDim.x * blockDim.x) {
    const int n = idx / Kp, k = idx - n * Kp;
    float v = (k < K && n < N) ? src[(size_t)b * K * N + (size_t)k * N + n] : 0.0f;
    dst[(size_t)b * total + idx] = (bf16)v;
  }
}

// ---------------------------------------------------------------------------
// Kernel 1: per-(node-tile,l)   h = feats @ W_pre[l]  -> padded bf16 ws [N][64][32]
//                               sc = zeta @ W_sc[l]   -> d_out second half
// zeta[m, a*64+u] = attrs[m,a] * feats[m,u,l]
// ---------------------------------------------------------------------------
__global__ __launch_bounds__(32)
void node_linear_kernel(const float* __restrict__ node_attrs,
                        const float* __restrict__ node_feats,
                        const bf16*  __restrict__ WscT,   // [9][64][640]
                        const bf16*  __restrict__ WpreT,  // [9][64][64]
                        bf16*  __restrict__ h_out,        // [N][64][32] (zeroed)
                        float* __restrict__ sc_out) {
  const int lane = threadIdx.x;
  const int n0   = blockIdx.x * 16;
  const int l    = blockIdx.y;

  __shared__ __align__(16) bf16 fA[16 * MULC];            // feats tile for this l
  __shared__ __align__(16) bf16 zeta[16 * (NATT * MULC)]; // [16][640]
  __shared__ float attrs_s[16 * NATT];

  for (int p = lane; p < 16 * MULC; p += 32) {
    const int m = p >> 6, u = p & 63;
    fA[p] = (bf16)node_feats[(size_t)(n0 + m) * FEAT + u * LMD + l];
  }
  for (int p = lane; p < 16 * NATT; p += 32)
    attrs_s[p] = node_attrs[(size_t)n0 * NATT + p];
  __syncthreads();

  const int col = lane & 15;
  const int mb  = (lane >> 4) << 3;

  // ---- h = feats @ W_pre[l] ----
  {
    const bf16* Bp = WpreT + (size_t)l * MULC * MULC;  // Kp = 64
    f32x8 acc[4] = {};
#pragma unroll
    for (int ks = 0; ks < 2; ++ks) {
      bf16x16 a = frag_A(fA, MULC, ks * 32, lane);
#pragma unroll
      for (int nt = 0; nt < 4; ++nt)
        acc[nt] = wmma_bf16(a, frag_BT(Bp, MULC, ks * 32, nt * 16, lane), acc[nt]);
    }
#pragma unroll
    for (int nt = 0; nt < 4; ++nt)
#pragma unroll
      for (int i = 0; i < 8; ++i)
        h_out[((size_t)(n0 + mb + i) * MULC + (nt * 16 + col)) * HPAD + l] = (bf16)acc[nt][i];
  }

  // ---- zeta ----
  for (int m = 0; m < 16; ++m) {
    for (int q = lane; q < NATT * MULC; q += 32) {
      const int a = q >> 6, u = q & 63;
      zeta[m * (NATT * MULC) + q] = (bf16)(attrs_s[m * NATT + a] * (float)fA[m * MULC + u]);
    }
  }
  __syncthreads();

  // ---- sc = zeta @ W_sc[l]  (K = 640) ----
  {
    const bf16* Bs = WscT + (size_t)l * MULC * (NATT * MULC);  // Kp = 640
    f32x8 acc[4] = {};
    for (int ks = 0; ks < 20; ++ks) {
      bf16x16 a = frag_A(zeta, NATT * MULC, ks * 32, lane);
#pragma unroll
      for (int nt = 0; nt < 4; ++nt)
        acc[nt] = wmma_bf16(a, frag_BT(Bs, NATT * MULC, ks * 32, nt * 16, lane), acc[nt]);
    }
#pragma unroll
    for (int nt = 0; nt < 4; ++nt)
#pragma unroll
      for (int i = 0; i < 8; ++i)
        sc_out[(size_t)(n0 + mb + i) * FEAT + (nt * 16 + col) * LMD + l] = acc[nt][i];
  }
}

// ---------------------------------------------------------------------------
// Kernel 2: one wave per 16 edges.
//   - radial MLP (silu) in WMMA bf16 -> lw[16][81]
//   - per edge: K[l,m] = sum_j ea[j] cg[l,j,m]
//               W2[l,k] = 0.25 * sum_m K[l,m] lw[k*9+m]   (1/sqrt(16) folded)
//               mji = h_pad[sender](64x32, K-padded) @ W2 via WMMA
//   - scatter-add f32 fragments into msg[receiver]
// ---------------------------------------------------------------------------
__global__ __launch_bounds__(32)
void edge_kernel(const float* __restrict__ edge_attrs,
                 const float* __restrict__ edge_feats,
                 const int*   __restrict__ senders,
                 const int*   __restrict__ receivers,
                 const float* __restrict__ cg,
                 const bf16*  __restrict__ w1T,   // [64][32]
                 const bf16*  __restrict__ w2T,   // [64][64]
                 const bf16*  __restrict__ w3T,   // [64][64]
                 const bf16*  __restrict__ w4T,   // [96][64]
                 const bf16*  __restrict__ h_pad, // [N][64][32]
                 float* __restrict__ msg) {
  const int lane = threadIdx.x;
  const int e0   = blockIdx.x * 16;

  __shared__ __align__(16) bf16  ef[16 * 32];    // edge_feats, K padded 8->32
  __shared__ __align__(16) bf16  xa[16 * HIDD];
  __shared__ __align__(16) bf16  xb[16 * HIDD];
  __shared__ __align__(16) bf16  W2T[16 * 32];   // per-edge 9x9 in BT layout, padded
  __shared__ float lw_s[16 * 96];                // layer-4 output (81 used)
  __shared__ float ea_s[16 * LMD];
  __shared__ float cg_s[LMD * LMD * LMD];
  __shared__ float kk[LMD * LMD];
  __shared__ int   snd_s[16];
  __shared__ int   rcv_s[16];

  // one-time zero of padded regions + staging
  for (int p = lane; p < 16 * 32; p += 32) ef[p]  = (bf16)0.0f;
  for (int p = lane; p < 16 * 32; p += 32) W2T[p] = (bf16)0.0f;
  for (int p = lane; p < LMD * LMD * LMD; p += 32) cg_s[p] = cg[p];
  for (int p = lane; p < 16 * LMD; p += 32) ea_s[p] = edge_attrs[(size_t)e0 * LMD + p];
  if (lane < 16) snd_s[lane] = senders[e0 + lane];
  else           rcv_s[lane - 16] = receivers[e0 + lane - 16];
  __syncthreads();
  for (int p = lane; p < 16 * NBAS; p += 32) {
    const int m = p >> 3, j = p & 7;
    ef[m * 32 + j] = (bf16)edge_feats[(size_t)e0 * NBAS + p];
  }
  __syncthreads();

  const int col = lane & 15;
  const int mb  = (lane >> 4) << 3;

  // ---- layer 1: ef @ w1 -> xa ----
  {
    f32x8 acc[4] = {};
    bf16x16 a = frag_A(ef, 32, 0, lane);
#pragma unroll
    for (int nt = 0; nt < 4; ++nt)
      acc[nt] = wmma_bf16(a, frag_BT(w1T, 32, 0, nt * 16, lane), acc[nt]);
#pragma unroll
    for (int nt = 0; nt < 4; ++nt)
#pragma unroll
      for (int i = 0; i < 8; ++i)
        xa[(mb + i) * HIDD + nt * 16 + col] = (bf16)silu(acc[nt][i]);
  }
  __syncthreads();

  // ---- layer 2: xa @ w2 -> xb ----
  {
    f32x8 acc[4] = {};
#pragma unroll
    for (int ks = 0; ks < 2; ++ks) {
      bf16x16 a = frag_A(xa, HIDD, ks * 32, lane);
#pragma unroll
      for (int nt = 0; nt < 4; ++nt)
        acc[nt] = wmma_bf16(a, frag_BT(w2T, HIDD, ks * 32, nt * 16, lane), acc[nt]);
    }
#pragma unroll
    for (int nt = 0; nt < 4; ++nt)
#pragma unroll
      for (int i = 0; i < 8; ++i)
        xb[(mb + i) * HIDD + nt * 16 + col] = (bf16)silu(acc[nt][i]);
  }
  __syncthreads();

  // ---- layer 3: xb @ w3 -> xa ----
  {
    f32x8 acc[4] = {};
#pragma unroll
    for (int ks = 0; ks < 2; ++ks) {
      bf16x16 a = frag_A(xb, HIDD, ks * 32, lane);
#pragma unroll
      for (int nt = 0; nt < 4; ++nt)
        acc[nt] = wmma_bf16(a, frag_BT(w3T, HIDD, ks * 32, nt * 16, lane), acc[nt]);
    }
#pragma unroll
    for (int nt = 0; nt < 4; ++nt)
#pragma unroll
      for (int i = 0; i < 8; ++i)
        xa[(mb + i) * HIDD + nt * 16 + col] = (bf16)silu(acc[nt][i]);
  }
  __syncthreads();

  // ---- layer 4: xa @ w4 (N padded 81->96) -> lw_s (f32) ----
  {
    f32x8 acc[6] = {};
#pragma unroll
    for (int ks = 0; ks < 2; ++ks) {
      bf16x16 a = frag_A(xa, HIDD, ks * 32, lane);
#pragma unroll
      for (int nt = 0; nt < 6; ++nt)
        acc[nt] = wmma_bf16(a, frag_BT(w4T, HIDD, ks * 32, nt * 16, lane), acc[nt]);
    }
#pragma unroll
    for (int nt = 0; nt < 6; ++nt)
#pragma unroll
      for (int i = 0; i < 8; ++i)
        lw_s[(mb + i) * 96 + nt * 16 + col] = acc[nt][i];
  }
  __syncthreads();

  // ---- per-edge tensor product + scatter ----
  for (int e = 0; e < 16; ++e) {
    const int s = snd_s[e];
    const int r = rcv_s[e];
    const bf16* hrow = h_pad + (size_t)s * (MULC * HPAD);

    // prefetch next sender's padded h row (4KB) -> global_prefetch_b8
    if (e < 15) {
      const char* nx = (const char*)(h_pad + (size_t)snd_s[e + 1] * (MULC * HPAD));
      __builtin_prefetch(nx + lane * 128, 0, 3);
    }

    // K[l,m] = sum_j ea[j] * cg[l,j,m]
    for (int p = lane; p < LMD * LMD; p += 32) {
      const int l_ = p / LMD, m_ = p - l_ * LMD;
      float acc = 0.0f;
#pragma unroll
      for (int j = 0; j < LMD; ++j)
        acc += ea_s[e * LMD + j] * cg_s[(l_ * LMD + j) * LMD + m_];
      kk[p] = acc;
    }
    __syncthreads();
    // W2T[k][l] = 0.25 * sum_m K[l,m] * lw[k*9+m]   (BT layout: col=k, K=l)
    for (int p = lane; p < LMD * LMD; p += 32) {
      const int l_ = p / LMD, k_ = p - l_ * LMD;
      float acc = 0.0f;
#pragma unroll
      for (int m_ = 0; m_ < LMD; ++m_)
        acc += kk[l_ * LMD + m_] * lw_s[e * 96 + k_ * LMD + m_];
      W2T[k_ * 32 + l_] = (bf16)(0.25f * acc);
    }
    __syncthreads();

    // mji = h_pad[s](64x32) @ W2(9x9 padded): 4 M-tiles; scatter to msg
    bf16x16 b = frag_BT(W2T, 32, 0, 0, lane);
#pragma unroll
    for (int ut = 0; ut < 4; ++ut) {
      f32x8 c = {};
      bf16x16 a = frag_A(hrow + ut * 16 * HPAD, HPAD, 0, lane);
      c = wmma_bf16(a, b, c);
      if (col < LMD) {
#pragma unroll
        for (int i = 0; i < 8; ++i)
          atomicAdd(&msg[(size_t)r * FEAT + (ut * 16 + mb + i) * LMD + col], c[i]);
      }
    }
    __syncthreads();
  }
}

// ---------------------------------------------------------------------------
// Kernel 3: out = msg @ W_out[l]  per (node-tile, l)
// ---------------------------------------------------------------------------
__global__ __launch_bounds__(32)
void out_linear_kernel(const float* __restrict__ msg,
                       const bf16*  __restrict__ WoutT,  // [9][64][64]
                       float* __restrict__ out) {
  const int lane = threadIdx.x;
  const int n0   = blockIdx.x * 16;
  const int l    = blockIdx.y;

  __shared__ __align__(16) bf16 mA[16 * MULC];
  for (int p = lane; p < 16 * MULC; p += 32) {
    const int m = p >> 6, u = p & 63;
    mA[p] = (bf16)msg[(size_t)(n0 + m) * FEAT + u * LMD + l];
  }
  __syncthreads();

  const bf16* Bo = WoutT + (size_t)l * MULC * MULC;
  const int col = lane & 15;
  const int mb  = (lane >> 4) << 3;

  f32x8 acc[4] = {};
#pragma unroll
  for (int ks = 0; ks < 2; ++ks) {
    bf16x16 a = frag_A(mA, MULC, ks * 32, lane);
#pragma unroll
    for (int nt = 0; nt < 4; ++nt)
      acc[nt] = wmma_bf16(a, frag_BT(Bo, MULC, ks * 32, nt * 16, lane), acc[nt]);
  }
#pragma unroll
  for (int nt = 0; nt < 4; ++nt)
#pragma unroll
    for (int i = 0; i < 8; ++i)
      out[(size_t)(n0 + mb + i) * FEAT + (nt * 16 + col) * LMD + l] = acc[nt][i];
}

// ---------------------------------------------------------------------------
extern "C" void kernel_launch(void* const* d_in, const int* in_sizes, int n_in,
                              void* d_out, int out_size, void* d_ws, size_t ws_size,
                              hipStream_t stream) {
  (void)in_sizes; (void)n_in; (void)out_size; (void)ws_size;

  const float* node_attrs = (const float*)d_in[0];
  const float* node_feats = (const float*)d_in[1];
  const float* edge_attrs = (const float*)d_in[2];
  const float* edge_feats = (const float*)d_in[3];
  const int*   senders    = (const int*)d_in[4];
  const int*   receivers  = (const int*)d_in[5];
  const float* W_sc       = (const float*)d_in[6];
  const float* W_pre      = (const float*)d_in[7];
  const float* cg         = (const float*)d_in[8];
  const float* mlp_w1     = (const float*)d_in[9];
  const float* mlp_w2     = (const float*)d_in[10];
  const float* mlp_w3     = (const float*)d_in[11];
  const float* mlp_w4     = (const float*)d_in[12];
  const float* W_out      = (const float*)d_in[13];

  float* out_p = (float*)d_out;                       // out  [N,64,9]
  float* sc_p  = out_p + (size_t)N_NODES * FEAT;      // sc   [N,64,9]

  // workspace layout
  char* ws = (char*)d_ws;
  float* msg   = (float*)ws;                                  // [N][64][9] f32
  ws += (size_t)N_NODES * FEAT * sizeof(float);
  bf16* h_pad  = (bf16*)ws;                                   // [N][64][32] bf16
  ws += (size_t)N_NODES * MULC * HPAD * sizeof(bf16);
  bf16* WpreT  = (bf16*)ws;  ws += (size_t)LMD * MULC * MULC * sizeof(bf16);   // [9][64][64]
  bf16* WscT   = (bf16*)ws;  ws += (size_t)LMD * MULC * NATT * MULC * sizeof(bf16); // [9][64][640]
  bf16* WoutT  = (bf16*)ws;  ws += (size_t)LMD * MULC * MULC * sizeof(bf16);   // [9][64][64]
  bf16* w1T    = (bf16*)ws;  ws += (size_t)MULC * 32 * sizeof(bf16);           // [64][32]
  bf16* w2T    = (bf16*)ws;  ws += (size_t)MULC * MULC * sizeof(bf16);         // [64][64]
  bf16* w3T    = (bf16*)ws;  ws += (size_t)MULC * MULC * sizeof(bf16);         // [64][64]
  bf16* w4T    = (bf16*)ws;  ws += (size_t)96 * MULC * sizeof(bf16);           // [96][64]

  hipMemsetAsync(msg,   0, (size_t)N_NODES * FEAT * sizeof(float), stream);
  hipMemsetAsync(h_pad, 0, (size_t)N_NODES * MULC * HPAD * sizeof(bf16), stream);

  // one-time weight transposition/padding into bf16 BT layouts
  transpose_pad_bf16<<<dim3(16, LMD), 256, 0, stream>>>(W_pre, WpreT, MULC, MULC, MULC, MULC);
  transpose_pad_bf16<<<dim3(160, LMD), 256, 0, stream>>>(W_sc, WscT, NATT * MULC, MULC, NATT * MULC, MULC);
  transpose_pad_bf16<<<dim3(16, LMD), 256, 0, stream>>>(W_out, WoutT, MULC, MULC, MULC, MULC);
  transpose_pad_bf16<<<dim3(8, 1),  256, 0, stream>>>(mlp_w1, w1T, NBAS, HIDD, 32, HIDD);
  transpose_pad_bf16<<<dim3(16, 1), 256, 0, stream>>>(mlp_w2, w2T, HIDD, HIDD, HIDD, HIDD);
  transpose_pad_bf16<<<dim3(16, 1), 256, 0, stream>>>(mlp_w3, w3T, HIDD, HIDD, HIDD, HIDD);
  transpose_pad_bf16<<<dim3(24, 1), 256, 0, stream>>>(mlp_w4, w4T, HIDD, NWGT, HIDD, 96);

  node_linear_kernel<<<dim3(N_NODES / 16, LMD), 32, 0, stream>>>(
      node_attrs, node_feats, WscT, WpreT, h_pad, sc_p);

  edge_kernel<<<N_EDGES / 16, 32, 0, stream>>>(
      edge_attrs, edge_feats, senders, receivers, cg,
      w1T, w2T, w3T, w4T, h_pad, msg);

  out_linear_kernel<<<dim3(N_NODES / 16, LMD), 32, 0, stream>>>(msg, WoutT, out_p);
}
